// WormholeInterconnect_56762287784390
// MI455X (gfx1250) — matrix operand order repaired and checked by reading, not compiled, not run on hardware
//
#include <hip/hip_runtime.h>
#include <stdint.h>

// ---------------------------------------------------------------------------
// Problem constants
// ---------------------------------------------------------------------------
#define BB   8          // batch
#define NN   1024       // sequence
#define DD   512        // feature dim
#define ROWS 8192       // BB*NN
#define OUT_ELEMS (8ull * BB * NN * DD)   // 8 bands of [B,N,D] fp32

typedef __attribute__((ext_vector_type(16))) __bf16 v16bf;
typedef __attribute__((ext_vector_type(8)))  float  v8f;
typedef __attribute__((ext_vector_type(4)))  int    v4i;

union FragU { uint4 u[2]; v16bf v; };

// ---- CDNA5 async global->LDS copy (ASYNCcnt-tracked), with safe fallback ----
#if defined(__has_builtin)
#  if __has_builtin(__builtin_amdgcn_global_load_async_to_lds_b128)
#    define HAVE_ASYNC_LDS 1
#  endif
#endif
#ifndef HAVE_ASYNC_LDS
#  define HAVE_ASYNC_LDS 0
#endif

#define AS1 __attribute__((address_space(1)))
#define AS3 __attribute__((address_space(3)))

__device__ __forceinline__ void async_cp16(const void* g, void* l) {
#if HAVE_ASYNC_LDS
  // Param 1 is v4i in AS1 (per compiler diagnostic); LDS flat addresses map to
  // LDS by dropping the upper bits (ISA 10.2), so the low 32 bits of the
  // generic pointer are the LDS byte offset.
  __builtin_amdgcn_global_load_async_to_lds_b128(
      (AS1 v4i*)(uintptr_t)g, (AS3 v4i*)(uint32_t)(uintptr_t)l, 0, 0);
#else
  *(uint4*)l = *(const uint4*)g;
#endif
}

__device__ __forceinline__ void async_wait() {
#if HAVE_ASYNC_LDS
#  if defined(__has_builtin) && __has_builtin(__builtin_amdgcn_s_wait_asynccnt)
  __builtin_amdgcn_s_wait_asynccnt(0);
#  else
  asm volatile("s_wait_asynccnt 0x0" ::: "memory");
#  endif
#endif
}

__device__ __forceinline__ unsigned short f2bf(float x) {
  unsigned u = __float_as_uint(x);
  unsigned r = (u + 0x7FFFu + ((u >> 16) & 1u)) >> 16;   // RNE
  return (unsigned short)r;
}
__device__ __forceinline__ float bf2f(unsigned short s) {
  return __uint_as_float(((unsigned)s) << 16);
}
__device__ __forceinline__ void bf8tof(uint4 u, float* o) {
  o[0] = bf2f((unsigned short)(u.x & 0xFFFFu)); o[1] = bf2f((unsigned short)(u.x >> 16));
  o[2] = bf2f((unsigned short)(u.y & 0xFFFFu)); o[3] = bf2f((unsigned short)(u.y >> 16));
  o[4] = bf2f((unsigned short)(u.z & 0xFFFFu)); o[5] = bf2f((unsigned short)(u.z >> 16));
  o[6] = bf2f((unsigned short)(u.w & 0xFFFFu)); o[7] = bf2f((unsigned short)(u.w >> 16));
}

struct BandPtrs { const float* p[8]; };

// Per-band call tables (kv band, scale, entropy slot), derived from reference:
// comp pairs e0..e5, bridge(kv=3) e6..e11, temporal(kv=7) e12..e18.
__constant__ int   d_ncalls[8]   = {3,3,3,1,3,3,3,0};
__constant__ int   d_kv[8][3]    = {{6,3,7},{5,3,7},{4,3,7},{7,7,7},{2,3,7},{1,3,7},{0,3,7},{0,0,0}};
__constant__ int   d_ent[8][3]   = {{1,6,12},{3,7,13},{5,8,14},{15,15,15},{4,9,16},{2,10,17},{0,11,18},{0,0,0}};
__constant__ float d_scale[8][3] = {{1.f,.5f,.5f},{1.f,.5f,.5f},{1.f,.5f,.5f},{.5f,.5f,.5f},
                                    {1.f,.5f,.5f},{1.f,.5f,.5f},{1.f,.5f,.5f},{0.f,0.f,0.f}};

// ---------------------------------------------------------------------------
// Kernel 1: bands fp32 -> bf16 (row-major, feeds WMMA A-fragments)
// grid = (ROWS*DD/4/256, 8)
// ---------------------------------------------------------------------------
__global__ __launch_bounds__(256) void k_cvt_bands(BandPtrs bp, unsigned short* __restrict__ dst) {
  int band = blockIdx.y;
  size_t i = ((size_t)blockIdx.x * 256 + threadIdx.x) * 4;
  float4 f = *(const float4*)(bp.p[band] + i);
  unsigned r0 = f2bf(f.x), r1 = f2bf(f.y), r2 = f2bf(f.z), r3 = f2bf(f.w);
  uint2 pk; pk.x = r0 | (r1 << 16); pk.y = r2 | (r3 << 16);
  *(uint2*)(dst + (size_t)band * ROWS * DD + i) = pk;
}

// ---------------------------------------------------------------------------
// Kernel 2: weights fp32 [e][d] -> bf16 transposed Wt [d][e]
// p: 0..6 = qW[p], 7..14 = kW[p-7], 15..22 = vW[p-15].  grid = (1024, 23)
// ---------------------------------------------------------------------------
__global__ __launch_bounds__(256) void k_cvt_w(const float* __restrict__ qW,
                                               const float* __restrict__ kW,
                                               const float* __restrict__ vW,
                                               unsigned short* __restrict__ Wt) {
  int p = blockIdx.y;
  int o = blockIdx.x * 256 + threadIdx.x;     // 0..262143
  int d = o >> 9, e = o & 511;
  const float* src = (p < 7)  ? qW + (size_t)p * 262144
                   : (p < 15) ? kW + (size_t)(p - 7) * 262144
                              : vW + (size_t)(p - 15) * 262144;
  Wt[(size_t)p * 262144 + o] = f2bf(src[(size_t)e * 512 + d]);
}

// ---------------------------------------------------------------------------
// Kernel 3: 23 projection GEMMs  Out[8192,512] = X_bf16 @ Wt + bias
// Block: 256 thr (8 waves), out tile 128x128, k-step 32.
// Software pipeline: async global->LDS copies (ASYNCcnt) into a double buffer,
// issue tile kt+1 before computing tile kt.
// Q -> QBF row-major bf16; K -> KBFT transposed bf16 [kv][b][e][n]; V -> VBF.
// grid = (64*4, 23)
// ---------------------------------------------------------------------------
__global__ __launch_bounds__(256) void k_proj(const unsigned short* __restrict__ bandsBF,
                                              const unsigned short* __restrict__ WtBF,
                                              const float* __restrict__ qb,
                                              const float* __restrict__ kb,
                                              const float* __restrict__ vb,
                                              unsigned short* __restrict__ QBF,
                                              unsigned short* __restrict__ KBFT,
                                              unsigned short* __restrict__ VBF) {
  int p = blockIdx.y;
  int mt = blockIdx.x >> 2, nt = blockIdx.x & 3;
  int mbase = mt * 128, nbase = nt * 128;
  int band = (p < 7) ? p : (p < 15) ? p - 7 : p - 15;
  const unsigned short* A = bandsBF + (size_t)band * ROWS * DD;
  const unsigned short* W = WtBF + (size_t)p * DD * DD;      // [d][e]

  __shared__ unsigned short As[2][128][40];    // 128 x 32 used (+pad, 16B rows)
  __shared__ unsigned short Bs[2][32][136];    // 32 x 128 used (+pad)

  int tid = threadIdx.x, w = tid >> 5, lane = tid & 31;
  int half = (lane < 16) ? 0 : 8;
  v8f acc[8] = {};

  auto stage = [&](int kt, int bufi) {
    int r = tid >> 1, cs = (tid & 1) * 16;                     // A: 128x32
    const unsigned short* ga = A + (size_t)(mbase + r) * DD + kt * 32 + cs;
    async_cp16(ga,     &As[bufi][r][cs]);
    async_cp16(ga + 8, &As[bufi][r][cs + 8]);
    int r2 = tid >> 3, cs2 = (tid & 7) * 16;                   // B: 32x128
    const unsigned short* gb = W + (size_t)(kt * 32 + r2) * DD + nbase + cs2;
    async_cp16(gb,     &Bs[bufi][r2][cs2]);
    async_cp16(gb + 8, &Bs[bufi][r2][cs2 + 8]);
  };

  stage(0, 0);
  for (int kt = 0; kt < 16; ++kt) {
    int cur = kt & 1;
    async_wait();            // this wave's async copies (tile kt) landed in LDS
    __syncthreads();         // everyone's copies landed; prev tile reads done
    if (kt < 15) stage(kt + 1, cur ^ 1);

    FragU af;                                   // A: 16x32 bf16
    int ar = w * 16 + (lane & 15);
    af.u[0] = *(const uint4*)&As[cur][ar][half];
    af.u[1] = *(const uint4*)&As[cur][ar][16 + half];
#pragma unroll
    for (int ct = 0; ct < 8; ++ct) {            // B: 32x16 bf16, lane = n
      FragU bf_;
      for (int s = 0; s < 16; ++s)
        ((unsigned short*)bf_.u)[s] = Bs[cur][(s < 8) ? half + s : 8 + half + s][ct * 16 + (lane & 15)];
      acc[ct] = __builtin_amdgcn_wmma_f32_16x16x32_bf16(
          false, af.v, false, bf_.v, (short)0, acc[ct], false, false);
    }
  }

  const float* bias = (p < 7)  ? qb + p * 512
                    : (p < 15) ? kb + (p - 7) * 512
                               : vb + (p - 15) * 512;
#pragma unroll
  for (int ct = 0; ct < 8; ++ct) {
    int e = nbase + ct * 16 + (lane & 15);
    float bv = bias[e];
#pragma unroll
    for (int r = 0; r < 8; ++r) {               // C/D layout: m = r + half, n = lane&15
      int m = mbase + w * 16 + r + half;
      unsigned short h = f2bf(acc[ct][r] + bv);
      if (p < 7) {
        QBF[(size_t)p * ROWS * DD + (size_t)m * DD + e] = h;
      } else if (p < 15) {
        int kv = p - 7, bbt = m >> 10, nq = m & 1023;
        KBFT[(((size_t)kv * 8 + bbt) * DD + e) * 1024 + nq] = h;
      } else {
        VBF[(size_t)(p - 15) * ROWS * DD + (size_t)m * DD + e] = h;
      }
    }
  }
}

// ---------------------------------------------------------------------------
// Kernel 4: per-row 1/max(||x||,eps) for Q (row-major) and K (transposed)
// grid = (7+8)*8192/8 blocks, 8 rows per 256-thr block (one wave per row)
// ---------------------------------------------------------------------------
__global__ __launch_bounds__(256) void k_rsq(const unsigned short* __restrict__ QBF,
                                             const unsigned short* __restrict__ KBFT,
                                             float* __restrict__ rsqQ,
                                             float* __restrict__ rsqK) {
  int g = blockIdx.x * 8 + (threadIdx.x >> 5);
  int lane = threadIdx.x & 31;
  float ss = 0.f;
  if (g < 7 * ROWS) {
    const unsigned short* row = QBF + (size_t)g * DD;
#pragma unroll
    for (int t = 0; t < 16; ++t) { float x = bf2f(row[lane + t * 32]); ss += x * x; }
  } else {
    int kk = g - 7 * ROWS;                       // kv*8192 + b*1024 + n
    int kv = kk >> 13, rem = kk & 8191, bbt = rem >> 10, n = rem & 1023;
    const unsigned short* base = KBFT + ((size_t)kv * 8 + bbt) * DD * 1024 + n;
#pragma unroll
    for (int t = 0; t < 16; ++t) { float x = bf2f(base[(size_t)(lane + t * 32) * 1024]); ss += x * x; }
  }
#pragma unroll
  for (int off = 16; off > 0; off >>= 1) ss += __shfl_xor(ss, off, 32);
  if (lane == 0) {
    float r = 1.f / fmaxf(sqrtf(ss), 1e-12f);
    if (g < 7 * ROWS) rsqQ[g] = r; else rsqK[g - 7 * ROWS] = r;
  }
}

// ---------------------------------------------------------------------------
// Kernel 5: per (band, batch, q-tile16) attention over that band's calls.
// Scores via WMMA into LDS, cosine scaling, wave top-8 + softmax + entropy,
// sparse attn@V gather into persistent register accumulator. No atomics.
// grid = (64, 8, 8), block = 256
// ---------------------------------------------------------------------------
__global__ __launch_bounds__(256) void k_attn(BandPtrs bp,
                                              const unsigned short* __restrict__ QBF,
                                              const unsigned short* __restrict__ KBFT,
                                              const unsigned short* __restrict__ VBF,
                                              const float* __restrict__ rsqQ,
                                              const float* __restrict__ rsqK,
                                              float* __restrict__ entStage,
                                              float* __restrict__ out) {
  int qt = blockIdx.x, b = blockIdx.y, band = blockIdx.z;
  int tid = threadIdx.x, w = tid >> 5, lane = tid & 31;
  int half = (lane < 16) ? 0 : 8;
  __shared__ float S[16][1024];                  // 64 KB score tile

  // Output accumulator: wave w owns local rows 2w,2w+1; lane owns 16 cols.
  float acc0[16], acc1[16];
  {
    const float* src = bp.p[band];
    size_t r0 = (size_t)(b * 1024 + qt * 16 + 2 * w) * 512 + lane * 16;
    const float4* g0 = (const float4*)(src + r0);
    const float4* g1 = (const float4*)(src + r0 + 512);
#pragma unroll
    for (int q = 0; q < 4; ++q) {
      float4 v0 = g0[q], v1 = g1[q];
      acc0[q*4+0] = v0.x; acc0[q*4+1] = v0.y; acc0[q*4+2] = v0.z; acc0[q*4+3] = v0.w;
      acc1[q*4+0] = v1.x; acc1[q*4+1] = v1.y; acc1[q*4+2] = v1.z; acc1[q*4+3] = v1.w;
    }
  }

  int nc = d_ncalls[band];
  for (int c = 0; c < nc; ++c) {
    int kv = d_kv[band][c];
    float scale = d_scale[band][c];
    int eidx = d_ent[band][c];

    // ---- Phase A: scores S[16][1024] = (Q tile) @ (K^T), wave w -> cols [128w,128w+128)
    const unsigned short* Qb = QBF + (size_t)band * ROWS * DD + (size_t)(b * 1024 + qt * 16) * DD;
    const unsigned short* Kb = KBFT + ((size_t)kv * 8 + b) * DD * 1024;   // [e][n]
    v8f sacc[8] = {};
    int arow = lane & 15;
    for (int kt = 0; kt < 16; ++kt) {
      FragU af;
      const unsigned short* ap = Qb + (size_t)arow * DD + kt * 32;
      af.u[0] = *(const uint4*)(ap + half);
      af.u[1] = *(const uint4*)(ap + 16 + half);
      const unsigned short* kp = Kb + (size_t)(kt * 32 + lane) * 1024 + w * 128;
      if (kt < 15) __builtin_prefetch(kp + 32 * 1024, 0, 1);   // next k-slab row
#pragma unroll
      for (int ct = 0; ct < 8; ++ct) {
        FragU bfm;
        for (int s = 0; s < 16; ++s)
          ((unsigned short*)bfm.u)[s] = Kb[(size_t)(kt * 32 + ((s < 8) ? half + s : 8 + half + s)) * 1024 + w * 128 + ct * 16 + (lane & 15)];
        sacc[ct] = __builtin_amdgcn_wmma_f32_16x16x32_bf16(
            false, af.v, false, bfm.v, (short)0, sacc[ct], false, false);
      }
    }
    float rq[8];
#pragma unroll
    for (int r = 0; r < 8; ++r) rq[r] = rsqQ[band * ROWS + b * 1024 + qt * 16 + r + half];
#pragma unroll
    for (int ct = 0; ct < 8; ++ct) {
      int nl = w * 128 + ct * 16 + (lane & 15);
      float rk = rsqK[kv * ROWS + b * 1024 + nl];
#pragma unroll
      for (int r = 0; r < 8; ++r) S[r + half][nl] = sacc[ct][r] * rq[r] * rk;
    }
    __syncthreads();

    // ---- Phase B: top-8 / softmax / entropy / sparse attn@V (2 rows per wave)
    for (int pass = 0; pass < 2; ++pass) {
      int lr = 2 * w + pass;
      float vals[32];
#pragma unroll
      for (int j = 0; j < 32; ++j) vals[j] = S[lr][lane + j * 32];
      unsigned dead = 0u;
      float tv[8]; int tc[8];
#pragma unroll
      for (int t = 0; t < 8; ++t) {
        float bv = -1e30f; int bc = 0x7fffffff;
#pragma unroll
        for (int j = 0; j < 32; ++j) {
          bool alive = ((dead >> j) & 1u) == 0u;
          float v = vals[j]; int col = lane + j * 32;
          if (alive && (v > bv || (v == bv && col < bc))) { bv = v; bc = col; }
        }
#pragma unroll
        for (int off = 16; off > 0; off >>= 1) {
          float ov = __shfl_xor(bv, off, 32);
          int   oc = __shfl_xor(bc, off, 32);
          if (ov > bv || (ov == bv && oc < bc)) { bv = ov; bc = oc; }
        }
        tv[t] = bv; tc[t] = bc;
        if ((bc & 31) == lane) dead |= (1u << (bc >> 5));
      }
      float mx = tv[0], Z = 0.f, a[8];
#pragma unroll
      for (int t = 0; t < 8; ++t) { a[t] = __expf(tv[t] - mx); Z += a[t]; }
      float inv = 1.f / Z, ent = 0.f;
#pragma unroll
      for (int t = 0; t < 8; ++t) { a[t] *= inv; ent -= a[t] * __logf(a[t] + 1e-9f); }
      if (lane == 0) entStage[(size_t)eidx * 8192 + b * 1024 + qt * 16 + lr] = ent;

      const unsigned short* Vb = VBF + (size_t)kv * ROWS * DD + (size_t)b * 1024 * DD + lane * 16;
      float* ac = pass ? acc1 : acc0;
#pragma unroll
      for (int t = 0; t < 8; ++t) {
        float wgt = scale * a[t];
        const uint4* vp = (const uint4*)(Vb + (size_t)tc[t] * DD);
        uint4 u0 = vp[0], u1 = vp[1];
        float f[16]; bf8tof(u0, f); bf8tof(u1, f + 8);
#pragma unroll
        for (int q2 = 0; q2 < 16; ++q2) ac[q2] += wgt * f[q2];
      }
    }
    __syncthreads();
  }

  { // store output tile
    size_t o0 = (((size_t)band * 8 + b) * 1024 + qt * 16 + 2 * w) * 512 + lane * 16;
    float4* g0 = (float4*)(out + o0);
    float4* g1 = (float4*)(out + o0 + 512);
#pragma unroll
    for (int q = 0; q < 4; ++q) {
      g0[q] = make_float4(acc0[q*4], acc0[q*4+1], acc0[q*4+2], acc0[q*4+3]);
      g1[q] = make_float4(acc1[q*4], acc1[q*4+1], acc1[q*4+2], acc1[q*4+3]);
    }
  }
}

// ---------------------------------------------------------------------------
// Kernel 6: entropy mean over Nq -> d_out tail [19][8].  grid = 19, block 256
// ---------------------------------------------------------------------------
__global__ __launch_bounds__(256) void k_entred(const float* __restrict__ stage,
                                                float* __restrict__ out) {
  int g = blockIdx.x * 8 + (threadIdx.x >> 5);   // (call*8 + b), 0..151
  int lane = threadIdx.x & 31;
  float s = 0.f;
#pragma unroll
  for (int t = 0; t < 32; ++t) s += stage[(size_t)g * 1024 + lane + t * 32];
#pragma unroll
  for (int off = 16; off > 0; off >>= 1) s += __shfl_xor(s, off, 32);
  if (lane == 0) out[OUT_ELEMS + g] = s * (1.f / 1024.f);
}

// ---------------------------------------------------------------------------
// Host launcher
// ---------------------------------------------------------------------------
extern "C" void kernel_launch(void* const* d_in, const int* in_sizes, int n_in,
                              void* d_out, int out_size, void* d_ws, size_t ws_size,
                              hipStream_t stream) {
  (void)in_sizes; (void)n_in; (void)out_size; (void)ws_size;

  BandPtrs bp;
  for (int i = 0; i < 8; ++i) bp.p[i] = (const float*)d_in[i];
  const float* qW = (const float*)d_in[8];
  const float* qb = (const float*)d_in[9];
  const float* kW = (const float*)d_in[10];
  const float* kb = (const float*)d_in[11];
  const float* vW = (const float*)d_in[12];
  const float* vb = (const float*)d_in[13];
  float* out = (float*)d_out;

  auto alignup = [](size_t x) { return (x + 255) & ~(size_t)255; };
  char* base = (char*)d_ws;
  unsigned short* bandsBF = (unsigned short*)base; base += alignup(8ull * ROWS * DD * 2);
  unsigned short* WtBF    = (unsigned short*)base; base += alignup(23ull * DD * DD * 2);
  unsigned short* QBF     = (unsigned short*)base; base += alignup(7ull * ROWS * DD * 2);
  unsigned short* KBFT    = (unsigned short*)base; base += alignup(8ull * 8 * DD * 1024 * 2);
  unsigned short* VBF     = (unsigned short*)base; base += alignup(8ull * ROWS * DD * 2);
  float* rsqQ     = (float*)base; base += alignup(7ull * ROWS * 4);
  float* rsqK     = (float*)base; base += alignup(8ull * ROWS * 4);
  float* entStage = (float*)base; base += alignup(19ull * 8192 * 4);

  k_cvt_bands<<<dim3(4096, 8), 256, 0, stream>>>(bp, bandsBF);
  k_cvt_w<<<dim3(1024, 23), 256, 0, stream>>>(qW, kW, vW, WtBF);
  k_proj<<<dim3(256, 23), 256, 0, stream>>>(bandsBF, WtBF, qb, kb, vb, QBF, KBFT, VBF);
  k_rsq<<<dim3(15360), 256, 0, stream>>>(QBF, KBFT, rsqQ, rsqK);
  k_attn<<<dim3(64, 8, 8), 256, 0, stream>>>(bp, QBF, KBFT, VBF, rsqQ, rsqK, entStage, out);
  k_entred<<<dim3(19), 256, 0, stream>>>(entStage, out);
}
